// GCHALayer_60885456388565
// MI455X (gfx1250) — matrix operand order repaired and checked by reading, not compiled, not run on hardware
//
#include <hip/hip_runtime.h>
#include <hip/hip_bf16.h>
#include <math.h>

#define D_MODEL   512
#define N_TOTAL   1024
#define HW_       4096
#define NUM_HEADS 8
#define D_K       64
#define BATCH     4
#define EPS2      0.04f      // EPSILON^2, dist<eps  <=>  dist2<eps2
#define SCORE_SCALE 0.125f   // 1/sqrt(64)

typedef __attribute__((ext_vector_type(16))) _Float16 v16h;
typedef __attribute__((ext_vector_type(8)))  _Float16 v8h;
typedef __attribute__((ext_vector_type(8)))  float    v8f;

// ---------------------------------------------------------------------------
// Fragment helpers (CDNA5 WMMA f16 16x16x32 layouts, CDNA5 ISA)
// A (16x32 f16): lane L holds row M=L&15; element e -> K = 16*(L>>4) + e (code's own K order, not ISA table)
//   => two adjacent 16B chunks at kbase and kbase+8, kbase = 16*(L>>4).
// B (32x16 f16): lane L holds col N=L&15; element e -> K = 16*(L>>4) + e (code's own K order, not ISA table)
//   => one contiguous 32B chunk.
// C/D (16x16 f32): lane L holds col N=L&15; element v -> row M = 8*(L>>4)+v.
// ---------------------------------------------------------------------------
__device__ __forceinline__ v16h load_a_f16(const _Float16* p) {
  v8h lo = *reinterpret_cast<const v8h*>(p);
  v8h hi = *reinterpret_cast<const v8h*>(p + 8);
  return __builtin_shufflevector(lo, hi, 0,1,2,3,4,5,6,7,8,9,10,11,12,13,14,15);
}

__device__ __forceinline__ v8f wmma_f16(v16h a, v16h b, v8f c) {
  return __builtin_amdgcn_wmma_f32_16x16x32_f16(false, a, false, b, (short)0, c,
                                                false, false);
}

// ---------------------------------------------------------------------------
// f32 -> f16 conversion
// ---------------------------------------------------------------------------
__global__ void cvt_f32_to_f16(const float* __restrict__ src,
                               _Float16* __restrict__ dst, int n) {
  int i = blockIdx.x * blockDim.x + threadIdx.x;
  if (i < n) dst[i] = (_Float16)src[i];
}

// ---------------------------------------------------------------------------
// Transposed geometric mask: maskT[m*N + n] = (|x_tilde[m]-x_hat[n,m]|^2 < eps^2)
// LDS tile transpose: coalesced x_hat reads (along m) and maskT writes (along n).
// ---------------------------------------------------------------------------
__global__ void geo_maskT_kernel(const float* __restrict__ xt,
                                 const float* __restrict__ xh,
                                 unsigned char* __restrict__ maskT) {
  __shared__ unsigned char tile[64][65];
  int n0 = (blockIdx.x & 15) * 64;   // N/64 = 16
  int m0 = (blockIdx.x >> 4) * 64;   // HW/64 = 64
  int tc = threadIdx.x & 63;
  int tr = threadIdx.x >> 6;         // 0..3
#pragma unroll
  for (int i = 0; i < 16; ++i) {
    int nl = tr + i * 4;
    int n = n0 + nl, m = m0 + tc;
    size_t idx = (size_t)n * HW_ + m;
    float dx = xt[2 * m + 0] - xh[2 * idx + 0];
    float dy = xt[2 * m + 1] - xh[2 * idx + 1];
    tile[nl][tc] = (dx * dx + dy * dy < EPS2) ? 1 : 0;
  }
  __syncthreads();
#pragma unroll
  for (int i = 0; i < 16; ++i) {
    int ml = tr + i * 4;
    maskT[(size_t)(m0 + ml) * N_TOTAL + n0 + tc] = tile[tc][ml];
  }
}

// ---------------------------------------------------------------------------
// Generic projection GEMM: Y = X @ W^T + b   (X: rows x 512 f16, W: 512x512 f16)
// One wave computes a 16(row) x 64(col) strip; A-fragment reused over 4 N-tiles.
// MODE 0: f16 out, per-head row-major  [b,h,pos,dk]      (Q, K)
// MODE 1: f16 out, per-head transposed [b,h,dk,pos]      (V)
// MODE 2: f32 out, plain row-major rows x 512            (final out proj)
// lbShift = log2(positions per batch); power-of-two, so no integer division.
// ---------------------------------------------------------------------------
template <int MODE>
__global__ void wmma_proj_kernel(const _Float16* __restrict__ X,
                                 const _Float16* __restrict__ W,
                                 const float* __restrict__ bias,
                                 void* __restrict__ dst,
                                 int rows, int lbShift) {
  int wave = (blockIdx.x * blockDim.x + threadIdx.x) >> 5;
  int lane = threadIdx.x & 31;
  int nwaves = (rows >> 4) * (D_MODEL / 64);
  if (wave >= nwaves) return;

  int row_tile = wave >> 3;
  int cb = (wave & 7) * 64;
  int half = lane >> 4;
  int l15  = lane & 15;
  int r = row_tile * 16 + l15;

  v8f acc[4] = {};
  const _Float16* xrow = X + (size_t)r * D_MODEL;
#pragma unroll 4
  for (int ks = 0; ks < D_MODEL / 32; ++ks) {
    v16h afrag = load_a_f16(xrow + ks * 32 + half * 16);
#pragma unroll
    for (int t = 0; t < 4; ++t) {
      int n = cb + t * 16 + l15;
      v16h bfrag = *reinterpret_cast<const v16h*>(
          W + (size_t)n * D_MODEL + ks * 32 + half * 16);
      acc[t] = wmma_f16(afrag, bfrag, acc[t]);
    }
  }

  int lbMask = (1 << lbShift) - 1;
#pragma unroll
  for (int t = 0; t < 4; ++t) {
    int c = cb + t * 16 + l15;
    float bv = bias[c];
#pragma unroll
    for (int v = 0; v < 8; ++v) {
      int m = row_tile * 16 + half * 8 + v;
      float y = acc[t][v] + bv;
      if (MODE == 2) {
        ((float*)dst)[(size_t)m * D_MODEL + c] = y;
      } else {
        int b = m >> lbShift, pos = m & lbMask;
        int h = c >> 6, d = c & 63;
        if (MODE == 0)
          ((_Float16*)dst)[(((size_t)(b * NUM_HEADS + h) << lbShift) + pos) * D_K + d] =
              (_Float16)y;
        else
          ((_Float16*)dst)[((size_t)(b * NUM_HEADS + h) * D_K + d) * HW_ + pos] =
              (_Float16)y;
      }
    }
  }
}

// ---------------------------------------------------------------------------
// Fused attention: scores -> mask -> exp -> rowsum -> attn write -> ctx GEMM.
// One workgroup (8 waves) owns (bh, 16-row block). The 16x4096 exp-tile lives
// in LDS as f16 (~131KB of the 320KB WGP budget). Softmax uses no max-pass
// (logits bounded, fp32 exp cannot overflow); normalization folded into the
// attn store and the ctx epilogue. All reductions are fixed-order.
// Masking is branchless: ev = maskf * __expf(s)  (mask byte is 0/1).
// ---------------------------------------------------------------------------
__global__ __launch_bounds__(256) void fused_attn_kernel(
    const _Float16* __restrict__ Qh, const _Float16* __restrict__ Kh,
    const _Float16* __restrict__ Vt, const unsigned char* __restrict__ maskT,
    float* __restrict__ attn, _Float16* __restrict__ ctx16) {
  __shared__ _Float16 eLDS[16][HW_ + 8];   // 131,328 B
  __shared__ float redbuf[8 * 16];
  __shared__ float invbuf[16];

  int bh    = blockIdx.x >> 6;   // B*H = 32
  int ntile = blockIdx.x & 63;   // N/16
  int w    = threadIdx.x >> 5;   // wave 0..7
  int lane = threadIdx.x & 31;
  int half = lane >> 4, l15 = lane & 15;

  // Q A-fragments (row = M = l15 within the 16-row block), loaded once.
  int r = ntile * 16 + l15;
  const _Float16* qrow = Qh + ((size_t)bh * N_TOTAL + r) * D_K + half * 16;
  v16h a0 = load_a_f16(qrow);        // K = 0..31
  v16h a1 = load_a_f16(qrow + 32);   // K = 32..63

  // ---- Phase 1: scores + exp into LDS, per-lane partial row sums ----------
  float sums[8] = {0.f, 0.f, 0.f, 0.f, 0.f, 0.f, 0.f, 0.f};
  for (int s8 = 0; s8 < 8; ++s8) {         // this wave's 8 column strips of 64
    int mbase = (w * 8 + s8) * 64;
#pragma unroll
    for (int t = 0; t < 4; ++t) {
      int m = mbase + t * 16 + l15;
      const _Float16* krow = Kh + ((size_t)bh * HW_ + m) * D_K + half * 16;
      v16h b0 = *reinterpret_cast<const v16h*>(krow);
      v16h b1 = *reinterpret_cast<const v16h*>(krow + 32);
      v8f sacc = {};
      sacc = wmma_f16(a0, b0, sacc);
      sacc = wmma_f16(a1, b1, sacc);
      // 8 mask bytes (0/1) for rows ntile*16 + half*8 + {0..7}
      const unsigned* mp = reinterpret_cast<const unsigned*>(
          maskT + (size_t)m * N_TOTAL + ntile * 16 + half * 8);
      unsigned mlo = mp[0], mhi = mp[1];
#pragma unroll
      for (int v = 0; v < 8; ++v) {
        unsigned mb = (v < 4) ? ((mlo >> (8 * v)) & 1u)
                              : ((mhi >> (8 * (v - 4))) & 1u);
        float ev = (float)mb * __expf(sacc[v] * SCORE_SCALE);
        sums[v] += ev;
        eLDS[half * 8 + v][m] = (_Float16)ev;
      }
    }
  }

  // ---- Phase 2: deterministic row-sum reduction ---------------------------
#pragma unroll
  for (int v = 0; v < 8; ++v) {
    float s = sums[v];
    s += __shfl_xor(s, 1);
    s += __shfl_xor(s, 2);
    s += __shfl_xor(s, 4);
    s += __shfl_xor(s, 8);   // xor<16 stays within the half
    sums[v] = s;
  }
  if (l15 == 0) {
#pragma unroll
    for (int v = 0; v < 8; ++v) redbuf[w * 16 + half * 8 + v] = sums[v];
  }
  __syncthreads();
  if (threadIdx.x < 16) {
    float tot = 0.f;
#pragma unroll
    for (int ww = 0; ww < 8; ++ww) tot += redbuf[ww * 16 + (int)threadIdx.x];
    invbuf[threadIdx.x] = 1.0f / tot;
  }
  __syncthreads();

  // ---- Phase 3: ctx partial WMMA from LDS (this wave's 512 K-columns) -----
  v8f acc[4] = {};
  for (int ks = 0; ks < 16; ++ks) {
    int kcol = w * 512 + ks * 32;
    v16h ae = load_a_f16(&eLDS[l15][kcol + half * 16]);
#pragma unroll
    for (int t = 0; t < 4; ++t) {
      int d = t * 16 + l15;
      v16h b = *reinterpret_cast<const v16h*>(
          Vt + ((size_t)bh * D_K + d) * HW_ + kcol + half * 16);
      acc[t] = wmma_f16(ae, b, acc[t]);
    }
  }

  // ---- Phase 4: write normalized attn (this wave's 16 x 512 region) ------
  {
    float* abase = attn + ((size_t)bh * N_TOTAL + ntile * 16) * HW_ + w * 512;
    for (int row = 0; row < 16; ++row) {
      float inv = invbuf[row];
      const _Float16* erow = &eLDS[row][w * 512];
      float* arow = abase + (size_t)row * HW_;
#pragma unroll
      for (int i = 0; i < 4; ++i) {
        int c = lane * 4 + i * 128;
        float4 o;
        o.x = (float)erow[c + 0] * inv;
        o.y = (float)erow[c + 1] * inv;
        o.z = (float)erow[c + 2] * inv;
        o.w = (float)erow[c + 3] * inv;
        *reinterpret_cast<float4*>(arow + c) = o;
      }
    }
  }
  __syncthreads();   // all eLDS reads done; safe to overlay

  // ---- Phase 5: cross-wave ctx reduction (LDS overlay, fixed order) ------
  float* ctxbuf = reinterpret_cast<float*>(&eLDS[0][0]);  // 8*16*64 f32 = 32KB
#pragma unroll
  for (int t = 0; t < 4; ++t)
#pragma unroll
    for (int v = 0; v < 8; ++v)
      ctxbuf[w * 1024 + (half * 8 + v) * 64 + t * 16 + l15] = acc[t][v];
  __syncthreads();

  int bb = bh >> 3, h = bh & 7;
  for (int idx = threadIdx.x; idx < 1024; idx += 256) {
    int row = idx >> 6, col = idx & 63;
    float s = 0.f;
#pragma unroll
    for (int ww = 0; ww < 8; ++ww) s += ctxbuf[ww * 1024 + idx];
    float val = s * invbuf[row];
    ctx16[((size_t)(bb * N_TOTAL + ntile * 16 + row)) * D_MODEL + h * D_K + col] =
        (_Float16)val;
  }
}

// ---------------------------------------------------------------------------
// Host launcher
// ---------------------------------------------------------------------------
extern "C" void kernel_launch(void* const* d_in, const int* in_sizes, int n_in,
                              void* d_out, int out_size, void* d_ws, size_t ws_size,
                              hipStream_t stream) {
  const float* q   = (const float*)d_in[0];
  const float* k   = (const float*)d_in[1];
  const float* v   = (const float*)d_in[2];
  const float* xt  = (const float*)d_in[3];
  const float* xh  = (const float*)d_in[4];
  const float* Wq  = (const float*)d_in[5];
  const float* bq  = (const float*)d_in[6];
  const float* Wk  = (const float*)d_in[7];
  const float* bk  = (const float*)d_in[8];
  const float* Wv  = (const float*)d_in[9];
  const float* bv  = (const float*)d_in[10];
  const float* Wo  = (const float*)d_in[11];
  const float* bo  = (const float*)d_in[12];

  float* out  = (float*)d_out;                               // (B, N, 512)
  float* attn = out + (size_t)BATCH * N_TOTAL * D_MODEL;     // (B, H, N, HW)

  char* ws = (char*)d_ws;
  size_t off = 0;
  auto take = [&](size_t bytes) {
    char* p = ws + off;
    off = (off + bytes + 255) & ~(size_t)255;
    return p;
  };
  const size_t NQ = (size_t)BATCH * N_TOTAL * D_MODEL;   // 2M
  const size_t NK = (size_t)BATCH * HW_ * D_MODEL;       // 8M
  _Float16* q16  = (_Float16*)take(NQ * 2);
  _Float16* k16  = (_Float16*)take(NK * 2);
  _Float16* v16  = (_Float16*)take(NK * 2);
  _Float16* wq16 = (_Float16*)take((size_t)D_MODEL * D_MODEL * 2);
  _Float16* wk16 = (_Float16*)take((size_t)D_MODEL * D_MODEL * 2);
  _Float16* wv16 = (_Float16*)take((size_t)D_MODEL * D_MODEL * 2);
  _Float16* wo16 = (_Float16*)take((size_t)D_MODEL * D_MODEL * 2);
  _Float16* Qh   = (_Float16*)take(NQ * 2);  // [b,h,N,64]
  _Float16* Kh   = (_Float16*)take(NK * 2);  // [b,h,HW,64]
  _Float16* Vt   = (_Float16*)take(NK * 2);  // [b,h,64,HW]
  _Float16* ctx  = (_Float16*)take(NQ * 2);  // [b,N,512]
  unsigned char* maskT = (unsigned char*)take((size_t)N_TOTAL * HW_); // [m][n]

  const int T = 256;
  // 1) f32 -> f16 conversions
  cvt_f32_to_f16<<<(int)((NQ + T - 1) / T), T, 0, stream>>>(q, q16, (int)NQ);
  cvt_f32_to_f16<<<(int)((NK + T - 1) / T), T, 0, stream>>>(k, k16, (int)NK);
  cvt_f32_to_f16<<<(int)((NK + T - 1) / T), T, 0, stream>>>(v, v16, (int)NK);
  int nw = D_MODEL * D_MODEL;
  cvt_f32_to_f16<<<(nw + T - 1) / T, T, 0, stream>>>(Wq, wq16, nw);
  cvt_f32_to_f16<<<(nw + T - 1) / T, T, 0, stream>>>(Wk, wk16, nw);
  cvt_f32_to_f16<<<(nw + T - 1) / T, T, 0, stream>>>(Wv, wv16, nw);
  cvt_f32_to_f16<<<(nw + T - 1) / T, T, 0, stream>>>(Wo, wo16, nw);

  // 2) transposed geometric mask
  geo_maskT_kernel<<<(N_TOTAL / 64) * (HW_ / 64), T, 0, stream>>>(xt, xh, maskT);

  // 3) projections (8 waves / block); lbShift: log2(positions per batch)
  {
    int waves = (BATCH * N_TOTAL / 16) * (D_MODEL / 64);   // 2048
    wmma_proj_kernel<0><<<waves / 8, T, 0, stream>>>(q16, wq16, bq, Qh,
                                                     BATCH * N_TOTAL, 10);
  }
  {
    int waves = (BATCH * HW_ / 16) * (D_MODEL / 64);       // 8192
    wmma_proj_kernel<0><<<waves / 8, T, 0, stream>>>(k16, wk16, bk, Kh,
                                                     BATCH * HW_, 12);
    wmma_proj_kernel<1><<<waves / 8, T, 0, stream>>>(v16, wv16, bv, Vt,
                                                     BATCH * HW_, 12);
  }

  // 4) fused scores + softmax + attn write + ctx GEMM
  fused_attn_kernel<<<BATCH * NUM_HEADS * (N_TOTAL / 16), T, 0, stream>>>(
      Qh, Kh, Vt, maskT, attn, ctx);

  // 5) output projection -> f32 out
  {
    int waves = (BATCH * N_TOTAL / 16) * (D_MODEL / 64);   // 2048
    wmma_proj_kernel<2><<<waves / 8, T, 0, stream>>>(ctx, wo16, bo, out,
                                                     BATCH * N_TOTAL, 10);
  }
}
